// GlobalAttentionPooling_33861522162212
// MI455X (gfx1250) — matrix operand order repaired and checked by reading, not compiled, not run on hardware
//
#include <hip/hip_runtime.h>
#include <hip/hip_bf16.h>

typedef __attribute__((ext_vector_type(16))) _Float16 v16h;
typedef __attribute__((ext_vector_type(8)))  _Float16 v8h;
typedef __attribute__((ext_vector_type(8)))  float    v8f;
typedef __attribute__((ext_vector_type(4)))  float    v4f;

#define INPUT_DIM_   256
#define ATTN_DIM_    64
#define NUM_GRAPHS_  512

// ---------------------------------------------------------------------------
// Kernel 1: per-node attention scores via WMMA.
//   logits = tanh(x @ W_att^T + b_att); scores = logits @ ctx
// Each wave handles a tile of 16 nodes. K=256 split into 8 steps of 32.
// A (16x32 f16): lane L holds row M=L&15; element j holds K=(j<8?j:j+8)+8*(L>>4)
// B (32x16 f16): lane L holds col N=L&15; element j holds K=(j<8?j:j+8)+8*(L>>4)
// D (16x16 f32): VGPR r, lane L -> M=r+8*(L>>4), N=L&15
// ---------------------------------------------------------------------------
__global__ __launch_bounds__(256) void attn_scores_kernel(
    const float* __restrict__ x, const float* __restrict__ W_att,
    const float* __restrict__ b_att, const float* __restrict__ ctx,
    float* __restrict__ scores, int N)
{
  __shared__ _Float16 sW[ATTN_DIM_ * INPUT_DIM_];  // 32 KB
  const int tid = threadIdx.x;
  for (int i = tid; i < ATTN_DIM_ * INPUT_DIM_; i += blockDim.x)
    sW[i] = (_Float16)W_att[i];
  __syncthreads();

  const int lane    = tid & 31;
  const int lane_lo = lane & 15;
  const int lane_hi = lane >> 4;
  const int wave    = tid >> 5;
  const int waves_per_blk = blockDim.x >> 5;
  const int wave_gid      = blockIdx.x * waves_per_blk + wave;
  const int nwaves        = gridDim.x * waves_per_blk;

  float ctxv[4], bav[4];
#pragma unroll
  for (int nt = 0; nt < 4; ++nt) {
    const int n = lane_lo + 16 * nt;
    ctxv[nt] = ctx[n];
    bav[nt]  = b_att[n];
  }

  const int ntiles = (N + 15) >> 4;
  for (int tile = wave_gid; tile < ntiles; tile += nwaves) {
    int nodeA = tile * 16 + lane_lo;
    if (nodeA >= N) nodeA = N - 1;  // tail clamp (stores are guarded)
    const float* __restrict__ xr = x + (size_t)nodeA * INPUT_DIM_;

    v8f acc[4];
#pragma unroll
    for (int nt = 0; nt < 4; ++nt) acc[nt] = (v8f){};

#pragma unroll
    for (int kk = 0; kk < 8; ++kk) {
      const int kbase = kk * 32;
      // ---- A fragment: 16 f32 -> f16, two 32B runs per lane ----
      const v4f* p0 = (const v4f*)(xr + kbase + 8 * lane_hi);
      const v4f* p1 = (const v4f*)(xr + kbase + 16 + 8 * lane_hi);
      v4f f0 = p0[0], f1 = p0[1];
      v4f f2 = p1[0], f3 = p1[1];
      v16h a;
#pragma unroll
      for (int j = 0; j < 4; ++j) {
        a[j]      = (_Float16)f0[j];
        a[j + 4]  = (_Float16)f1[j];
        a[j + 8]  = (_Float16)f2[j];
        a[j + 12] = (_Float16)f3[j];
      }
      // ---- B fragments from LDS, one WMMA per 16-wide N tile ----
#pragma unroll
      for (int nt = 0; nt < 4; ++nt) {
        const int n = lane_lo + 16 * nt;
        const _Float16* bp = &sW[n * INPUT_DIM_ + kbase + 8 * lane_hi];
        v8h b0 = *(const v8h*)(bp);
        v8h b1 = *(const v8h*)(bp + 16);
        v16h b = __builtin_shufflevector(b0, b1, 0, 1, 2, 3, 4, 5, 6, 7,
                                         8, 9, 10, 11, 12, 13, 14, 15);
        acc[nt] = __builtin_amdgcn_wmma_f32_16x16x32_f16(
            false, a, false, b, (short)0, acc[nt], false, false);
      }
    }

    // ---- score[m] = sum_n tanh(logit + b) * ctx ----
    float tsum[8];
#pragma unroll
    for (int r = 0; r < 8; ++r) tsum[r] = 0.f;
#pragma unroll
    for (int nt = 0; nt < 4; ++nt) {
#pragma unroll
      for (int r = 0; r < 8; ++r)
        tsum[r] += tanhf(acc[nt][r] + bav[nt]) * ctxv[nt];
    }
    // reduce across the 16 lanes of each half (n dimension)
#pragma unroll
    for (int r = 0; r < 8; ++r) {
      float v = tsum[r];
      v += __shfl_xor(v, 1, 32);
      v += __shfl_xor(v, 2, 32);
      v += __shfl_xor(v, 4, 32);
      v += __shfl_xor(v, 8, 32);
      tsum[r] = v;
    }
    float my = 0.f;
#pragma unroll
    for (int r = 0; r < 8; ++r) my = (lane_lo == r) ? tsum[r] : my;
    const int nodeo = tile * 16 + lane_lo + 8 * lane_hi;  // M = lane_lo + 8*lane_hi
    if (lane_lo < 8 && nodeo < N) scores[nodeo] = my;
  }
}

// ---------------------------------------------------------------------------
// Kernel 2: segment boundaries from sorted batch ids -> seg_start[G+1]
// ---------------------------------------------------------------------------
__global__ void seg_bounds_kernel(const int* __restrict__ batch,
                                  int* __restrict__ seg_start, int N, int G)
{
  const int i = blockIdx.x * blockDim.x + threadIdx.x;
  if (i >= N) return;
  const int b = batch[i];
  if (i == 0) {
    for (int g = 0; g <= b; ++g) seg_start[g] = 0;
  } else {
    const int pb = batch[i - 1];
    if (pb != b)
      for (int g = pb + 1; g <= b; ++g) seg_start[g] = i;
  }
  if (i == N - 1)
    for (int g = b + 1; g <= G; ++g) seg_start[g] = N;
}

// ---------------------------------------------------------------------------
// Kernel 3: block-per-graph stable softmax + weighted pooling.
// thread d owns output column d (coalesced x reads).
// ---------------------------------------------------------------------------
__global__ __launch_bounds__(256) void pool_kernel(
    const float* __restrict__ x, const float* __restrict__ scores,
    const int* __restrict__ seg_start, float* __restrict__ pooled)
{
  const int g   = blockIdx.x;
  const int tid = threadIdx.x;
  const int s = seg_start[g];
  const int e = seg_start[g + 1];

  __shared__ float red[256];
  __shared__ float wsh[256];

  // segment max
  float m = -3.402823466e38f;
  for (int i = s + tid; i < e; i += 256) m = fmaxf(m, scores[i]);
  red[tid] = m;
  __syncthreads();
  for (int o = 128; o > 0; o >>= 1) {
    if (tid < o) red[tid] = fmaxf(red[tid], red[tid + o]);
    __syncthreads();
  }
  m = red[0];
  __syncthreads();

  // segment sum of exp
  float ssum = 0.f;
  for (int i = s + tid; i < e; i += 256) ssum += expf(scores[i] - m);
  red[tid] = ssum;
  __syncthreads();
  for (int o = 128; o > 0; o >>= 1) {
    if (tid < o) red[tid] += red[tid + o];
    __syncthreads();
  }
  const float inv = 1.f / (red[0] + 1e-8f);
  __syncthreads();

  // weighted accumulation: acc_d = sum_i exp(s_i - m) * x[i][d]
  float acc = 0.f;
  for (int base = s; base < e; base += 256) {
    const int cnt = min(256, e - base);
    __syncthreads();
    if (tid < cnt) wsh[tid] = expf(scores[base + tid] - m);
    __syncthreads();
    int j = 0;
    for (; j + 8 <= cnt; j += 8) {
#pragma unroll
      for (int u = 0; u < 8; ++u)
        acc += wsh[j + u] * x[(size_t)(base + j + u) * INPUT_DIM_ + tid];
    }
    for (; j < cnt; ++j)
      acc += wsh[j] * x[(size_t)(base + j) * INPUT_DIM_ + tid];
  }
  pooled[(size_t)g * INPUT_DIM_ + tid] = acc * inv;
}

// ---------------------------------------------------------------------------
// Kernel 4: out = pooled @ W_out^T + b_out   (512 x 256, K=256; tiny)
// ---------------------------------------------------------------------------
__global__ __launch_bounds__(256) void out_mm_kernel(
    const float* __restrict__ pooled, const float* __restrict__ W_out,
    const float* __restrict__ b_out, float* __restrict__ out)
{
  const int g = blockIdx.x;
  const int o = threadIdx.x;
  __shared__ float p[INPUT_DIM_];
  p[o] = pooled[(size_t)g * INPUT_DIM_ + o];
  __syncthreads();
  float acc = b_out[o];
  const float* __restrict__ w = W_out + (size_t)o * INPUT_DIM_;
#pragma unroll 8
  for (int d = 0; d < INPUT_DIM_; ++d) acc += p[d] * w[d];
  out[(size_t)g * INPUT_DIM_ + o] = acc;
}

// ---------------------------------------------------------------------------
extern "C" void kernel_launch(void* const* d_in, const int* in_sizes, int n_in,
                              void* d_out, int out_size, void* d_ws, size_t ws_size,
                              hipStream_t stream)
{
  const float* x     = (const float*)d_in[0];
  const int*   batch = (const int*)d_in[1];
  const float* W_att = (const float*)d_in[2];
  const float* b_att = (const float*)d_in[3];
  const float* ctx   = (const float*)d_in[4];
  const float* W_out = (const float*)d_in[5];
  const float* b_out = (const float*)d_in[6];
  float* out = (float*)d_out;

  const int N = in_sizes[1];          // number of nodes (batch length)
  const int G = NUM_GRAPHS_;

  // workspace layout
  size_t off = 0;
  float* scores = (float*)((char*)d_ws + off);
  off += ((size_t)N * sizeof(float) + 255) & ~(size_t)255;
  int* seg_start = (int*)((char*)d_ws + off);
  off += ((size_t)(G + 1) * sizeof(int) + 255) & ~(size_t)255;
  float* pooled = (float*)((char*)d_ws + off);

  // 1. attention scores (WMMA)
  attn_scores_kernel<<<1024, 256, 0, stream>>>(x, W_att, b_att, ctx, scores, N);
  // 2. segment boundaries
  seg_bounds_kernel<<<(N + 255) / 256, 256, 0, stream>>>(batch, seg_start, N, G);
  // 3. softmax + weighted pooling, block per graph
  pool_kernel<<<G, 256, 0, stream>>>(x, scores, seg_start, pooled);
  // 4. output projection
  out_mm_kernel<<<G, 256, 0, stream>>>(pooled, W_out, b_out, out);
}
